// SinkhornAttention_83141976917055
// MI455X (gfx1250) — matrix operand order repaired and checked by reading, not compiled, not run on hardware
//
#include <hip/hip_runtime.h>

// ---------------- problem constants ----------------
constexpr int kBS  = 4;
constexpr int kSEQ = 4096;
constexpr int kH   = 16;
constexpr int kHD  = 64;
constexpr int kBLK = 128;
constexpr int kNB  = 32;
constexpr size_t kTensorElems = (size_t)kBS * kSEQ * 1024; // 16,777,216 per tensor
constexpr size_t kTensorBytesH = kTensorElems * 2;         // f16 storage

// ---------------- vector types ----------------
typedef _Float16 h2   __attribute__((ext_vector_type(2)));
typedef _Float16 h4   __attribute__((ext_vector_type(4)));
typedef _Float16 h8   __attribute__((ext_vector_type(8)));
typedef _Float16 v16h __attribute__((ext_vector_type(16)));
typedef float    v8f  __attribute__((ext_vector_type(8)));
typedef int      v4i  __attribute__((ext_vector_type(4)));

#define WMMA_F16(a, b, c) \
  __builtin_amdgcn_wmma_f32_16x16x32_f16(false, (a), false, (b), (short)0, (c), false, false)

// ---------------- async global->LDS (CDNA5), guarded ----------------
#ifndef __has_builtin
#define __has_builtin(x) 0
#endif
#if __has_builtin(__builtin_amdgcn_global_load_async_to_lds_b128)
#define HAVE_ASYNC 1
#else
#define HAVE_ASYNC 0
#endif

#if HAVE_ASYNC
#define ASG __attribute__((address_space(1)))
#define ASL __attribute__((address_space(3)))
__device__ __forceinline__ void cp_async16(const _Float16* g, _Float16* l) {
  // signature (from hipcc diagnostic): (v4i __device__* src, v4i __shared__* dst, int, int)
  __builtin_amdgcn_global_load_async_to_lds_b128((ASG v4i*)g, (ASL v4i*)l, 0, 0);
}
__device__ __forceinline__ void async_wait() {
#if __has_builtin(__builtin_amdgcn_s_wait_asynccnt)
  __builtin_amdgcn_s_wait_asynccnt(0);
#else
  asm volatile("s_wait_asynccnt 0x0" ::: "memory");
#endif
}
#else
__device__ __forceinline__ void cp_async16(const _Float16* g, _Float16* l) {
  *(h8*)l = *(const h8*)g;
}
__device__ __forceinline__ void async_wait() {}
#endif

// ---------------- fragment helpers ----------------
__device__ __forceinline__ v16h frag_h(const _Float16* plo, const _Float16* phi) {
  h8 lo = *(const h8*)plo;
  h8 hi = *(const h8*)phi;
  return __builtin_shufflevector(lo, hi, 0,1,2,3,4,5,6,7,8,9,10,11,12,13,14,15);
}
__device__ __forceinline__ v16h frag_f(const float* plo, const float* phi) {
  float4 a0 = ((const float4*)plo)[0], a1 = ((const float4*)plo)[1];
  float4 b0 = ((const float4*)phi)[0], b1 = ((const float4*)phi)[1];
  v16h r;
  r[0]=(_Float16)a0.x;  r[1]=(_Float16)a0.y;  r[2]=(_Float16)a0.z;  r[3]=(_Float16)a0.w;
  r[4]=(_Float16)a1.x;  r[5]=(_Float16)a1.y;  r[6]=(_Float16)a1.z;  r[7]=(_Float16)a1.w;
  r[8]=(_Float16)b0.x;  r[9]=(_Float16)b0.y;  r[10]=(_Float16)b0.z; r[11]=(_Float16)b0.w;
  r[12]=(_Float16)b1.x; r[13]=(_Float16)b1.y; r[14]=(_Float16)b1.z; r[15]=(_Float16)b1.w;
  return r;
}
__device__ __forceinline__ h8 pack8(float4 a, float4 b) {
  h8 r;
  r[0]=(_Float16)a.x; r[1]=(_Float16)a.y; r[2]=(_Float16)a.z; r[3]=(_Float16)a.w;
  r[4]=(_Float16)b.x; r[5]=(_Float16)b.y; r[6]=(_Float16)b.z; r[7]=(_Float16)b.w;
  return r;
}

// =====================================================================
// Dense GEMM  out[16384 x 1024] = A[16384 x 1024] * W[1024 x 1024]
// A_HALF: A tensor stored f16 (raw copy staging); else f32 (cvt staging).
// OUT_HALF: store f16 (intermediate tensors); else f32 (final output).
// 128x128 tile, 8 waves, K-tile 64, software-pipelined global prefetch.
// =====================================================================
template <bool A_HALF, bool OUT_HALF>
__global__ __launch_bounds__(256)
void gemm_bias_kernel(const void* __restrict__ Ap, const float* __restrict__ W,
                      const float* __restrict__ bias, void* __restrict__ outp,
                      float outScale) {
  __shared__ __align__(16) _Float16 sA[128][72];
  __shared__ __align__(16) _Float16 sB[128][72];

  const int t    = threadIdx.x;
  const int wv   = t >> 5;
  const int lane = t & 31;
  const int g    = lane >> 4;
  const int ln   = lane & 15;
  const int m0   = blockIdx.y * 128;
  const int n0   = blockIdx.x * 128;

  const int arow = t >> 1, akc = (t & 1) * 32;
  const int bk0  = (t >> 5) * 8;
  const int bn0  = (t & 31) * 4;

  const float*    Af = (const float*)Ap;
  const _Float16* Ah = (const _Float16*)Ap;

  float4 ar[8];   // f32 A staging regs
  h8     arh[4];  // f16 A staging regs
  float4 br[8];   // W staging regs (8 k-rows x 4 n)

  auto load_tile = [&](int kt, float4* arr, h8* arrh, float4* brr) {
    if constexpr (A_HALF) {
      const _Float16* asrc = Ah + (size_t)(m0 + arow) * 1024 + kt + akc;
#pragma unroll
      for (int i = 0; i < 4; ++i) arrh[i] = ((const h8*)asrc)[i];
    } else {
      const float* asrc = Af + (size_t)(m0 + arow) * 1024 + kt + akc;
#pragma unroll
      for (int i = 0; i < 8; ++i) arr[i] = ((const float4*)asrc)[i];
    }
#pragma unroll
    for (int i = 0; i < 8; ++i)
      brr[i] = *(const float4*)(W + (size_t)(kt + bk0 + i) * 1024 + n0 + bn0);
  };
  auto store_tile = [&](const float4* arr, const h8* arrh, const float4* brr) {
    if constexpr (A_HALF) {
#pragma unroll
      for (int i = 0; i < 4; ++i) *(h8*)&sA[arow][akc + i * 8] = arrh[i];
    } else {
#pragma unroll
      for (int i = 0; i < 4; ++i)
        *(h8*)&sA[arow][akc + i * 8] = pack8(arr[i * 2], arr[i * 2 + 1]);
    }
#pragma unroll
    for (int half = 0; half < 2; ++half) {
      const float* r0 = (const float*)&brr[half * 4 + 0];
      const float* r1 = (const float*)&brr[half * 4 + 1];
      const float* r2 = (const float*)&brr[half * 4 + 2];
      const float* r3 = (const float*)&brr[half * 4 + 3];
#pragma unroll
      for (int j = 0; j < 4; ++j) {
        h4 hv;
        hv[0] = (_Float16)r0[j]; hv[1] = (_Float16)r1[j];
        hv[2] = (_Float16)r2[j]; hv[3] = (_Float16)r3[j];
        *(h4*)&sB[bn0 + j][bk0 + half * 4] = hv;
      }
    }
  };

  v8f c[8] = {};
  load_tile(0, ar, arh, br);

  for (int kt = 0; kt < 1024; kt += 64) {
    store_tile(ar, arh, br);
    __syncthreads();

    float4 ar2[8]; h8 arh2[4]; float4 br2[8];
    const bool more = (kt + 64) < 1024;
    if (more) load_tile(kt + 64, ar2, arh2, br2);  // overlaps WMMA below

#pragma unroll
    for (int kb = 0; kb < 64; kb += 32) {
      const _Float16* ap = &sA[wv * 16 + ln][kb];
      v16h af = frag_h(ap + 8 * g, ap + 16 + 8 * g);
      v16h bf[8];
#pragma unroll
      for (int nt = 0; nt < 8; ++nt) {
        const _Float16* bp = &sB[nt * 16 + ln][kb + 8 * g];
        bf[nt] = frag_h(bp, bp + 16);
      }
#pragma unroll
      for (int nt = 0; nt < 8; ++nt) c[nt] = WMMA_F16(af, bf[nt], c[nt]);
    }
    __syncthreads();

    if (more) {
#pragma unroll
      for (int i = 0; i < 8; ++i) ar[i] = ar2[i];
#pragma unroll
      for (int i = 0; i < 4; ++i) arh[i] = arh2[i];
#pragma unroll
      for (int i = 0; i < 8; ++i) br[i] = br2[i];
    }
  }

#pragma unroll
  for (int nt = 0; nt < 8; ++nt) {
    const int n = n0 + nt * 16 + ln;
    const float bv = bias[n];
#pragma unroll
    for (int r = 0; r < 8; ++r) {
      const int m = m0 + wv * 16 + g * 8 + r;
      const float val = (c[nt][r] + bv) * outScale;
      if constexpr (OUT_HALF)
        ((_Float16*)outp)[(size_t)m * 1024 + n] = (_Float16)val;
      else
        ((float*)outp)[(size_t)m * 1024 + n] = val;
    }
  }
}

// =====================================================================
// sum_key over BLK rows (f16 source, f32 accumulate) + sort logits.
// =====================================================================
__global__ __launch_bounds__(256)
void sumkey_sort_kernel(const _Float16* __restrict__ k, const float* __restrict__ w_sort,
                        const float* __restrict__ b_sort, float* __restrict__ sort_out) {
  const int b = blockIdx.x >> 5, n = blockIdx.x & 31;
  const int t = threadIdx.x;
  __shared__ float sacc[1024];
  __shared__ float part[8][32];

  float a0 = 0.f, a1 = 0.f, a2 = 0.f, a3 = 0.f;
  const _Float16* base = k + (size_t)(b * kSEQ + n * kBLK) * 1024 + t * 4;
  for (int s = 0; s < kBLK; ++s) {
    h4 vv = *(const h4*)(base + (size_t)s * 1024);
    a0 += (float)vv[0]; a1 += (float)vv[1]; a2 += (float)vv[2]; a3 += (float)vv[3];
  }
  sacc[t * 4 + 0] = a0; sacc[t * 4 + 1] = a1;
  sacc[t * 4 + 2] = a2; sacc[t * 4 + 3] = a3;
  __syncthreads();

  const int m = t & 31, seg = t >> 5;
  float p = 0.f;
  for (int i = seg * 128; i < seg * 128 + 128; ++i)
    p += sacc[i] * w_sort[(size_t)i * 32 + m];
  part[seg][m] = p;
  __syncthreads();

  if (t < 32) {
    float tot = b_sort[t];
#pragma unroll
    for (int s = 0; s < 8; ++s) tot += part[s][t];
    sort_out[((size_t)b * kNB + n) * 32 + t] = tot;
  }
}

// =====================================================================
// Sinkhorn normalization, one wave per batch (row per lane).
// =====================================================================
__global__ __launch_bounds__(32)
void sinkhorn_kernel(const float* __restrict__ sort_out, float* __restrict__ perm) {
  const int b = blockIdx.x;
  const int l = threadIdx.x;
  __shared__ float mat[32][33];
  __shared__ float clse[32];

  float vals[32];
#pragma unroll
  for (int j = 0; j < 32; ++j)
    vals[j] = sort_out[((size_t)b * 32 + l) * 32 + j];  // TEMP == 1.0

  for (int it = 0; it < 5; ++it) {
    float mx = vals[0];
#pragma unroll
    for (int j = 1; j < 32; ++j) mx = fmaxf(mx, vals[j]);
    float s = 0.f;
#pragma unroll
    for (int j = 0; j < 32; ++j) s += __expf(vals[j] - mx);
    const float rlse = mx + __logf(s);
#pragma unroll
    for (int j = 0; j < 32; ++j) { vals[j] -= rlse; mat[l][j] = vals[j]; }
    __syncthreads();
    float cm = mat[0][l];
    for (int i = 1; i < 32; ++i) cm = fmaxf(cm, mat[i][l]);
    float cs = 0.f;
    for (int i = 0; i < 32; ++i) cs += __expf(mat[i][l] - cm);
    clse[l] = cm + __logf(cs);
    __syncthreads();
#pragma unroll
    for (int j = 0; j < 32; ++j) vals[j] -= clse[j];
  }
#pragma unroll
  for (int j = 0; j < 32; ++j)
    perm[((size_t)b * 32 + l) * 32 + j] = __expf(fminf(fmaxf(vals[j], -1.f), 1.f));
}

// =====================================================================
// Block mixing (f16 in/out, f32 accumulate), 2 elements/thread via h2.
// =====================================================================
__global__ __launch_bounds__(256)
void blockmix_kernel(const float* __restrict__ perm, const _Float16* __restrict__ k,
                     const _Float16* __restrict__ v, _Float16* __restrict__ sk,
                     _Float16* __restrict__ sv) {
  const int b = blockIdx.y;
  const _Float16* src = blockIdx.z ? v : k;
  _Float16* dst = blockIdx.z ? sv : sk;
  __shared__ float sp[32][32];
  const int t = threadIdx.x;
  for (int i = t; i < 1024; i += 256) sp[i >> 5][i & 31] = perm[(size_t)b * 1024 + i];
  __syncthreads();

  const size_t inner = ((size_t)blockIdx.x * 256 + t) * 2;
  const _Float16* sbase = src + (size_t)b * 4194304 + inner;
  h2 vals[32];
#pragma unroll 8
  for (int m = 0; m < 32; ++m) vals[m] = *(const h2*)(sbase + (size_t)m * 131072);
  _Float16* dbase = dst + (size_t)b * 4194304 + inner;
  for (int n = 0; n < 32; ++n) {
    float a0 = 0.f, a1 = 0.f;
#pragma unroll
    for (int m = 0; m < 32; ++m) {
      a0 += sp[n][m] * (float)vals[m][0];
      a1 += sp[n][m] * (float)vals[m][1];
    }
    h2 o; o[0] = (_Float16)a0; o[1] = (_Float16)a1;
    *(h2*)(dbase + (size_t)n * 131072) = o;
  }
}

// =====================================================================
// Block attention: one block = (b, 64-row half of a 128-block, h).
// Q/K tiles staged via GLOBAL_LOAD_ASYNC_TO_LDS_B128 (f16, raw copy);
// V^T staged via 4x4 register repack; WMMA logits -> exp -> WMMA PV.
// =====================================================================
constexpr int SQ_OFF  = 0;
constexpr int SK_OFF  = SQ_OFF + 64 * 72 * 2;      //  9216
constexpr int SVT_OFF = SK_OFF + 256 * 72 * 2;     // 46080
constexpr int SL_OFF  = SVT_OFF + 64 * 264 * 2;    // 79872
constexpr int RS_OFF  = SL_OFF + 64 * 264 * 4;     // 147456
constexpr int SMEM_BYTES = RS_OFF + 64 * 4;        // 147712

__global__ __launch_bounds__(128)
void attn_kernel(const _Float16* __restrict__ q, const _Float16* __restrict__ k,
                 const _Float16* __restrict__ v, const _Float16* __restrict__ sk,
                 const _Float16* __restrict__ sv, _Float16* __restrict__ x) {
  extern __shared__ char smem[];
  _Float16* sQ   = (_Float16*)(smem + SQ_OFF);   // [64][72]
  _Float16* sK   = (_Float16*)(smem + SK_OFF);   // [256][72]
  _Float16* sVt  = (_Float16*)(smem + SVT_OFF);  // [64][264]
  float*    sL   = (float*)(smem + SL_OFF);      // [64][264]
  float*    rsum = (float*)(smem + RS_OFF);      // [64]

  const int t = threadIdx.x;
  const int wv = t >> 5, lane = t & 31, g = lane >> 4, ln = lane & 15;
  const int nblk = blockIdx.x >> 1, half = blockIdx.x & 1;
  const int h = blockIdx.y, b = blockIdx.z;
  const int s0 = nblk * kBLK;
  const int q0 = s0 + half * 64;

  // ---- stage Q: async b128 copies (4 chunks per thread) ----
  {
    const int row = t >> 1, c0 = (t & 1) * 4;
    const _Float16* src = q + ((size_t)(b * kSEQ + q0 + row) * kH + h) * kHD + c0 * 8;
#pragma unroll
    for (int c = 0; c < 4; ++c)
      cp_async16(src + c * 8, &sQ[row * 72 + (c0 + c) * 8]);
  }
  // ---- stage Kcat rows: async b128 copies (2 rows x 8 chunks / thread) ----
#pragma unroll
  for (int rr = 0; rr < 2; ++rr) {
    const int kk = t * 2 + rr;
    const _Float16* src = (kk < kBLK)
        ? k + ((size_t)(b * kSEQ + s0 + kk) * kH + h) * kHD
        : sk + ((size_t)b * 4194304 + (size_t)nblk * 131072 +
                (size_t)(kk - kBLK) * 1024 + h * kHD);
#pragma unroll
    for (int c = 0; c < 8; ++c)
      cp_async16(src + c * 8, &sK[kk * 72 + c * 8]);
  }
  // ---- stage Vcat^T: 4 kk rows per thread, packed b64 stores ----
  {
    const int kk0 = (t & 63) * 4;
    const int d0  = (t >> 6) * 32;
    const _Float16* src[4];
#pragma unroll
    for (int i = 0; i < 4; ++i) {
      const int kk = kk0 + i;
      src[i] = (kk < kBLK)
          ? v + ((size_t)(b * kSEQ + s0 + kk) * kH + h) * kHD + d0
          : sv + ((size_t)b * 4194304 + (size_t)nblk * 131072 +
                  (size_t)(kk - kBLK) * 1024 + h * kHD) + d0;
    }
#pragma unroll
    for (int c = 0; c < 4; ++c) {  // chunks of 8 d
      h8 r0 = ((const h8*)src[0])[c];
      h8 r1 = ((const h8*)src[1])[c];
      h8 r2 = ((const h8*)src[2])[c];
      h8 r3 = ((const h8*)src[3])[c];
#pragma unroll
      for (int j = 0; j < 8; ++j) {
        h4 hv;
        hv[0] = r0[j]; hv[1] = r1[j]; hv[2] = r2[j]; hv[3] = r3[j];
        *(h4*)&sVt[(size_t)(d0 + c * 8 + j) * 264 + kk0] = hv;
      }
    }
  }
  async_wait();
  __syncthreads();

  // ---- GEMM1: logits (wave strip 16 x 256), K = 64 ----
  v8f c1[16] = {};
#pragma unroll
  for (int kb = 0; kb < 64; kb += 32) {
    const _Float16* ap = sQ + (wv * 16 + ln) * 72 + kb;
    v16h af = frag_h(ap + 8 * g, ap + 16 + 8 * g);
    v16h bf[16];
#pragma unroll
    for (int nt = 0; nt < 16; ++nt) {
      const _Float16* bp = sK + (nt * 16 + ln) * 72 + kb + 8 * g;
      bf[nt] = frag_h(bp, bp + 16);
    }
#pragma unroll
    for (int nt = 0; nt < 16; ++nt) c1[nt] = WMMA_F16(af, bf[nt], c1[nt]);
  }
#pragma unroll
  for (int nt = 0; nt < 16; ++nt)
#pragma unroll
    for (int r = 0; r < 8; ++r)
      sL[(wv * 16 + g * 8 + r) * 264 + nt * 16 + ln] = c1[nt][r];
  __syncthreads();

  // ---- softmax: unnormalized exp, deferred 1/rowsum ----
  if (t < 64) {
    float* row = sL + t * 264;
    float mx = row[0];
    for (int j = 1; j < 256; ++j) mx = fmaxf(mx, row[j]);
    float s = 0.f;
    for (int j = 0; j < 256; ++j) { float e = __expf(row[j] - mx); row[j] = e; s += e; }
    rsum[t] = s;
  }
  __syncthreads();

  // ---- GEMM2: out = P(64x256) * Vcat(256x64) ----
  v8f c2[4] = {};
  for (int kb = 0; kb < 256; kb += 32) {
    const float* apf = sL + (wv * 16 + ln) * 264 + kb;
    v16h af = frag_f(apf + 8 * g, apf + 16 + 8 * g);
    v16h bf[4];
#pragma unroll
    for (int nt = 0; nt < 4; ++nt) {
      const _Float16* bp = sVt + (nt * 16 + ln) * 264 + kb + 8 * g;
      bf[nt] = frag_h(bp, bp + 16);
    }
#pragma unroll
    for (int nt = 0; nt < 4; ++nt) c2[nt] = WMMA_F16(af, bf[nt], c2[nt]);
  }

  // ---- epilogue: normalize, write x (f16) ----
#pragma unroll
  for (int nt = 0; nt < 4; ++nt) {
    const int d = nt * 16 + ln;
#pragma unroll
    for (int r = 0; r < 8; ++r) {
      const int ml = wv * 16 + g * 8 + r;
      const float val = c2[nt][r] / rsum[ml];
      x[((size_t)(b * kSEQ + q0 + ml) * kH + h) * kHD + d] = (_Float16)val;
    }
  }
}

// =====================================================================
// Host launch
// =====================================================================
extern "C" void kernel_launch(void* const* d_in, const int* in_sizes, int n_in,
                              void* d_out, int out_size, void* d_ws, size_t ws_size,
                              hipStream_t stream) {
  (void)in_sizes; (void)n_in; (void)out_size; (void)ws_size;
  const float* inputs_q = (const float*)d_in[0];
  const float* wq = (const float*)d_in[1];
  const float* bq = (const float*)d_in[2];
  const float* wk = (const float*)d_in[3];
  const float* bk = (const float*)d_in[4];
  const float* wv = (const float*)d_in[5];
  const float* bv = (const float*)d_in[6];
  const float* w_sort = (const float*)d_in[7];
  const float* b_sort = (const float*)d_in[8];
  const float* wo = (const float*)d_in[9];
  const float* bo = (const float*)d_in[10];
  float* out = (float*)d_out;

  char* wsb = (char*)d_ws;
  _Float16* qh  = (_Float16*)(wsb + 0 * kTensorBytesH);
  _Float16* kh  = (_Float16*)(wsb + 1 * kTensorBytesH);
  _Float16* vh  = (_Float16*)(wsb + 2 * kTensorBytesH);
  _Float16* skh = (_Float16*)(wsb + 3 * kTensorBytesH);
  _Float16* svh = (_Float16*)(wsb + 4 * kTensorBytesH);
  _Float16* xh  = (_Float16*)(wsb + 5 * kTensorBytesH);
  float* sort_out = (float*)(wsb + 6 * kTensorBytesH);
  float* perm     = sort_out + 4096;

  const dim3 gg(8, 128);  // N/128, M/128
  gemm_bias_kernel<false, true><<<gg, 256, 0, stream>>>(inputs_q, wq, bq, qh, 0.125f);
  gemm_bias_kernel<false, true><<<gg, 256, 0, stream>>>(inputs_q, wk, bk, kh, 1.0f);
  gemm_bias_kernel<false, true><<<gg, 256, 0, stream>>>(inputs_q, wv, bv, vh, 1.0f);

  sumkey_sort_kernel<<<dim3(kBS * kNB), 256, 0, stream>>>(kh, w_sort, b_sort, sort_out);
  sinkhorn_kernel<<<dim3(kBS), 32, 0, stream>>>(sort_out, perm);
  blockmix_kernel<<<dim3(256, kBS, 2), 256, 0, stream>>>(perm, kh, vh, skh, svh);

  attn_kernel<<<dim3(kNB * 2, kH, kBS), 128, SMEM_BYTES, stream>>>(qh, kh, vh, skh, svh, xh);

  gemm_bias_kernel<true, false><<<gg, 256, 0, stream>>>(xh, wo, bo, out, 1.0f);
}
